// CrossAttention_52424370815682
// MI455X (gfx1250) — matrix operand order repaired and checked by reading, not compiled, not run on hardware
//
#include <hip/hip_runtime.h>

// MI455X / gfx1250 cross-attention. bf16 WMMA everywhere, fp32 accumulate.
// B=4, NQ=NK=1024, D=1024, H=16, HD=64.
// Roofline: ~52 GFLOP total vs ~80MB traffic -> matrix-core bound; everything
// lives in the 192MB L2 after first touch. Pipeline:
//   gemm_proj x3 (Q scaled by HD^-0.5*log2e; V stored transposed [B][D][NK])
//   flash attention (TDM-staged K/V tiles in LDS, DPP softmax reductions)
//   gemm_out (bf16 @ fp32W + bias -> fp32)
// Workspace: Qb(8MB) | Kb(8MB) | Vt(8MB) | AO(8MB) = 32MB bf16.

typedef __attribute__((ext_vector_type(16))) __bf16 v16bf;
typedef __attribute__((ext_vector_type(8)))  float  v8f;
typedef unsigned int u32x4 __attribute__((ext_vector_type(4)));
typedef int          i32x8 __attribute__((ext_vector_type(8)));
typedef int          i32x4 __attribute__((ext_vector_type(4)));

#define LDSPAD  40  // GEMM LDS row stride (bf16): 80B -> conflict-free
#define KSTRIDE 72  // attn K/V LDS row stride (bf16): 144B = 128B row + 16B TDM pad
#define PSTRIDE 72  // attn P LDS row stride

__device__ __forceinline__ v8f wmma_bf16(v16bf a, v16bf b, v8f c) {
  return __builtin_amdgcn_wmma_f32_16x16x32_bf16(false, a, false, b, (short)0, c,
                                                 false, false);
}

__device__ __forceinline__ void ld_frag32B(v16bf& v, const __bf16* p) {
  ((uint4*)&v)[0] = ((const uint4*)p)[0];
  ((uint4*)&v)[1] = ((const uint4*)p)[1];
}

// ---- DPP 16-lane butterfly reductions (no LDS, dual-issuable VALU) ----
template <int CTRL>
__device__ __forceinline__ float dpp_movf(float x) {
  return __builtin_bit_cast(
      float, __builtin_amdgcn_update_dpp(0, __builtin_bit_cast(int, x), CTRL,
                                         0xF, 0xF, true));
}
__device__ __forceinline__ float rowmax16(float x) {
  x = fmaxf(x, dpp_movf<0xB1>(x));   // quad_perm(1,0,3,2)  == xor 1
  x = fmaxf(x, dpp_movf<0x4E>(x));   // quad_perm(2,3,0,1)  == xor 2
  x = fmaxf(x, dpp_movf<0x141>(x));  // row_half_mirror     == xor 4 (post-fold)
  x = fmaxf(x, dpp_movf<0x140>(x));  // row_mirror          == xor 8 (post-fold)
  return x;
}
__device__ __forceinline__ float rowsum16(float x) {
  x += dpp_movf<0xB1>(x);
  x += dpp_movf<0x4E>(x);
  x += dpp_movf<0x141>(x);
  x += dpp_movf<0x140>(x);
  return x;
}

// ---------------------------------------------------------------------------
// GEMM: out = A(fp32 [4096][1024]) @ W(fp32 [1024][1024]) * oscale, out bf16.
// TRANS=false: out[m][n] row-major; TRANS=true: Vt[b][n][kpos], b=m>>10.
// ---------------------------------------------------------------------------
template <bool TRANS>
__global__ __launch_bounds__(256) void gemm_proj(const float* __restrict__ A,
                                                 const float* __restrict__ W,
                                                 __bf16* __restrict__ out,
                                                 float oscale) {
  __shared__ __bf16 As[128 * LDSPAD];  // As[row][k]
  __shared__ __bf16 Bs[128 * LDSPAD];  // Bs[n][k] (W transposed)

  const int tid = threadIdx.x, lane = tid & 31, wave = tid >> 5;
  const int m0 = blockIdx.y * 128, n0 = blockIdx.x * 128;
  const int wm = (wave & 3) * 32, wn = (wave >> 2) * 64;
  const int rb = lane & 15, hi = lane >> 4;
  const int koff8 = hi * 8, koff16 = hi * 16;
  const int ar = tid >> 1, ak = (tid & 1) * 16;
  const int wk = tid >> 3, wn2 = (tid & 7) * 16;

  v8f acc[2][4] = {};

  for (int kk = 0; kk < 1024; kk += 32) {
    {
      const float* ag = A + (size_t)(m0 + ar) * 1024 + kk + ak;
      float f[16];
#pragma unroll
      for (int j = 0; j < 4; ++j) {
        float4 t = ((const float4*)ag)[j];
        f[j * 4 + 0] = t.x; f[j * 4 + 1] = t.y; f[j * 4 + 2] = t.z; f[j * 4 + 3] = t.w;
      }
      __bf16* as = As + ar * LDSPAD + ak;
#pragma unroll
      for (int j = 0; j < 16; ++j) as[j] = (__bf16)f[j];
    }
    {
      const float* wg = W + (size_t)(kk + wk) * 1024 + n0 + wn2;
      float f[16];
#pragma unroll
      for (int j = 0; j < 4; ++j) {
        float4 t = ((const float4*)wg)[j];
        f[j * 4 + 0] = t.x; f[j * 4 + 1] = t.y; f[j * 4 + 2] = t.z; f[j * 4 + 3] = t.w;
      }
#pragma unroll
      for (int j = 0; j < 16; ++j) Bs[(wn2 + j) * LDSPAD + wk] = (__bf16)f[j];
    }
    __syncthreads();

    v16bf afr[2], bfr[4];
#pragma unroll
    for (int mi = 0; mi < 2; ++mi) {
      const __bf16* p = As + (wm + mi * 16 + rb) * LDSPAD;
      ((uint4*)&afr[mi])[0] = *(const uint4*)(p + koff16);
      ((uint4*)&afr[mi])[1] = *(const uint4*)(p + koff16 + 8);
    }
#pragma unroll
    for (int ni = 0; ni < 4; ++ni)
      ld_frag32B(bfr[ni], Bs + (wn + ni * 16 + rb) * LDSPAD + koff16);
#pragma unroll
    for (int mi = 0; mi < 2; ++mi)
#pragma unroll
      for (int ni = 0; ni < 4; ++ni)
        acc[mi][ni] = wmma_bf16(afr[mi], bfr[ni], acc[mi][ni]);
    __syncthreads();
  }

#pragma unroll
  for (int mi = 0; mi < 2; ++mi) {
#pragma unroll
    for (int ni = 0; ni < 4; ++ni) {
      const int mt = m0 + wm + mi * 16 + hi * 8;
      const int nt = n0 + wn + ni * 16 + rb;
      if (TRANS) {
        const int bidx = mt >> 10, kpos = mt & 1023;
        __bf16 tmp[8];
#pragma unroll
        for (int i = 0; i < 8; ++i) tmp[i] = (__bf16)(acc[mi][ni][i] * oscale);
        *(uint4*)(out + ((size_t)(bidx * 1024 + nt)) * 1024 + kpos) = *(uint4*)tmp;
      } else {
#pragma unroll
        for (int i = 0; i < 8; ++i)
          out[(size_t)(mt + i) * 1024 + nt] = (__bf16)(acc[mi][ni][i] * oscale);
      }
    }
  }
}

// ---------------------------------------------------------------------------
// Flash attention: block = 8 waves = 128 queries of one (b,h). kv-step 64.
// K/V tiles staged in LDS by the Tensor Data Mover (one issue per block),
// softmax row-reductions via DPP, P routed C-layout->LDS->A-layout.
// `stage_mode` (host passes 0) selects cooperative staging at runtime; the
// branch is never taken but keeps LDS stores visible so the compiler cannot
// fold the TDM-written tile loads, and Ks/Vs reads stay addrspace(3) ds ops.
// ---------------------------------------------------------------------------
__global__ __launch_bounds__(256) void attn_kernel(const __bf16* __restrict__ Qb,
                                                   const __bf16* __restrict__ Kb,
                                                   const __bf16* __restrict__ Vt,
                                                   __bf16* __restrict__ AO,
                                                   int stage_mode) {
  __shared__ __bf16 Ks[64 * KSTRIDE];       // Ks[kv][d]
  __shared__ __bf16 Vs[64 * KSTRIDE];       // Vs[d][kv]  (from transposed Vt)
  __shared__ __bf16 Pl[8 * 16 * PSTRIDE];   // per-wave P tile 16x64

  const int tid = threadIdx.x, lane = tid & 31, wave = tid >> 5;
  const int b = blockIdx.x >> 4, h = blockIdx.x & 15;
  const int q0 = blockIdx.y * 128 + wave * 16;
  const int rb = lane & 15, hi = lane >> 4;
  const int koff8 = hi * 8, koff16 = hi * 16;

  // Q fragments (already scaled by HD^-0.5 * log2(e) at projection time)
  const __bf16* qrow = Qb + ((size_t)(b * 1024 + q0 + rb)) * 1024 + h * 64;
  v16bf a0, a1;
  ((uint4*)&a0)[0] = *(const uint4*)(qrow + koff16);
  ((uint4*)&a0)[1] = *(const uint4*)(qrow + koff16 + 8);
  ((uint4*)&a1)[0] = *(const uint4*)(qrow + 32 + koff16);
  ((uint4*)&a1)[1] = *(const uint4*)(qrow + 32 + koff16 + 8);

  v8f o[4] = {};
  float mrun[8], lrun[8];
#pragma unroll
  for (int i = 0; i < 8; ++i) { mrun[i] = -3.0e38f; lrun[i] = 0.0f; }

  __bf16* pw = Pl + wave * 16 * PSTRIDE;

  for (int kv = 0; kv < 1024; kv += 64) {
    // ---- stage K (64x64) and V (64x64) tiles into LDS ----
    bool did_tdm = false;
#if __has_builtin(__builtin_amdgcn_tensor_load_to_lds)
    if (stage_mode == 0) {
      did_tdm = true;
      if (wave == 0) {
        // D# group1: mask=0, data_size=2B(code1), pad_enable, pad_interval=32dw,
        // pad_amount=4dw; tensor [4096][1024], stride 1024; tile 64x64.
        i32x8 g1;
        g1[0] = (1 << 16) | (1 << 20) | (4 << 22) | (3 << 25);
        g1[1] = (int)(1024u << 16);   // tensor_dim0 = 1024 (low 16 in [31:16])
        g1[2] = (int)(4096u << 16);   // tensor_dim1 = 4096 (low 16 in [31:16])
        g1[3] = (int)(64u << 16);     // tile_dim0 = 64
        g1[4] = 64;                   // tile_dim1 = 64, tile_dim2 = 0
        g1[5] = 1024;                 // tensor_dim0_stride
        g1[6] = 0;
        g1[7] = 0;
        i32x4 gz4 = {};               // groups 2/3 unused (2D tile)
        i32x8 gz8 = {};               // 6-arg form's extra group (unused)

        unsigned long long gaK =
            (unsigned long long)(uintptr_t)(Kb + ((size_t)(b * 1024 + kv)) * 1024 + h * 64);
        u32x4 g0;
        g0[0] = 1u;  // count=1
        g0[1] = (unsigned)(uintptr_t)(void*)Ks;
        g0[2] = (unsigned)gaK;
        g0[3] = (unsigned)((gaK >> 32) & 0x01FFFFFFull) | (2u << 30);  // type=2
        __builtin_amdgcn_tensor_load_to_lds(g0, g1, gz4, gz4, gz8, 0);

        unsigned long long gaV =
            (unsigned long long)(uintptr_t)(Vt + ((size_t)(b * 1024 + h * 64)) * 1024 + kv);
        g0[1] = (unsigned)(uintptr_t)(void*)Vs;
        g0[2] = (unsigned)gaV;
        g0[3] = (unsigned)((gaV >> 32) & 0x01FFFFFFull) | (2u << 30);
        __builtin_amdgcn_tensor_load_to_lds(g0, g1, gz4, gz4, gz8, 0);

        __builtin_amdgcn_s_wait_tensorcnt(0);
      }
    }
#endif
    if (!did_tdm) {
      // Cooperative staging (also keeps LDS stores visible to the optimizer).
      const int r = tid >> 2, c = (tid & 3) * 16;
      const __bf16* kg = Kb + ((size_t)(b * 1024 + kv + r)) * 1024 + h * 64 + c;
      *(uint4*)(Ks + r * KSTRIDE + c)     = ((const uint4*)kg)[0];
      *(uint4*)(Ks + r * KSTRIDE + c + 8) = ((const uint4*)kg)[1];
      const __bf16* vg = Vt + ((size_t)(b * 1024 + h * 64 + r)) * 1024 + kv + c;
      *(uint4*)(Vs + r * KSTRIDE + c)     = ((const uint4*)vg)[0];
      *(uint4*)(Vs + r * KSTRIDE + c + 8) = ((const uint4*)vg)[1];
    }
    __syncthreads();

    // ---- S = Q K^T : 4 column tiles of 16 ----
    v8f s[4];
#pragma unroll
    for (int ct = 0; ct < 4; ++ct) {
      const __bf16* kr = Ks + (ct * 16 + rb) * KSTRIDE + koff16;
      v16bf bk;
      v8f z = {};
      ld_frag32B(bk, kr);       z = wmma_bf16(a0, bk, z);
      ld_frag32B(bk, kr + 32);  z = wmma_bf16(a1, bk, z);
      s[ct] = z;
    }

    // ---- online softmax (exp2 domain; scale pre-folded into Q) ----
    float vm[8], corr[8], rs[8], p[4][8];
#pragma unroll
    for (int i = 0; i < 8; ++i) {
      vm[i] = fmaxf(fmaxf(s[0][i], s[1][i]), fmaxf(s[2][i], s[3][i]));
      vm[i] = rowmax16(vm[i]);
      float mn = fmaxf(mrun[i], vm[i]);
      corr[i] = exp2f(mrun[i] - mn);
      mrun[i] = mn;
    }
#pragma unroll
    for (int ct = 0; ct < 4; ++ct)
#pragma unroll
      for (int i = 0; i < 8; ++i) p[ct][i] = exp2f(s[ct][i] - mrun[i]);
#pragma unroll
    for (int i = 0; i < 8; ++i) {
      rs[i] = (p[0][i] + p[1][i]) + (p[2][i] + p[3][i]);
      rs[i] = rowsum16(rs[i]);
      lrun[i] = lrun[i] * corr[i] + rs[i];
    }
#pragma unroll
    for (int t = 0; t < 4; ++t)
#pragma unroll
      for (int i = 0; i < 8; ++i) o[t][i] = o[t][i] * corr[i];

    // ---- P: C-layout -> LDS -> A-layout (wave-private, DS in-order) ----
#pragma unroll
    for (int ct = 0; ct < 4; ++ct)
#pragma unroll
      for (int i = 0; i < 8; ++i)
        pw[(i + hi * 8) * PSTRIDE + ct * 16 + rb] = (__bf16)p[ct][i];
    asm volatile("" ::: "memory");
    v16bf pa0, pa1;
    const __bf16* pr = pw + rb * PSTRIDE;
    ((uint4*)&pa0)[0] = *(const uint4*)(pr + koff16);
    ((uint4*)&pa0)[1] = *(const uint4*)(pr + koff16 + 8);
    ((uint4*)&pa1)[0] = *(const uint4*)(pr + 32 + koff16);
    ((uint4*)&pa1)[1] = *(const uint4*)(pr + 32 + koff16 + 8);

    // ---- O += P * V ----
#pragma unroll
    for (int t = 0; t < 4; ++t) {
      const __bf16* vr = Vs + (t * 16 + rb) * KSTRIDE + koff16;
      v16bf bv;
      ld_frag32B(bv, vr);       o[t] = wmma_bf16(pa0, bv, o[t]);
      ld_frag32B(bv, vr + 32);  o[t] = wmma_bf16(pa1, bv, o[t]);
    }
    __syncthreads();  // protect LDS tiles before next stage
  }

  // ---- normalize + store ----
  float inv[8];
#pragma unroll
  for (int i = 0; i < 8; ++i) inv[i] = 1.0f / lrun[i];
#pragma unroll
  for (int t = 0; t < 4; ++t) {
    const int col = h * 64 + t * 16 + rb;
#pragma unroll
    for (int i = 0; i < 8; ++i)
      AO[((size_t)(b * 1024 + q0 + hi * 8 + i)) * 1024 + col] =
          (__bf16)(o[t][i] * inv[i]);
  }
}

// ---------------------------------------------------------------------------
// Output projection: d_out(fp32) = AO(bf16 [4096][1024]) @ Wo(fp32) + bo.
// ---------------------------------------------------------------------------
__global__ __launch_bounds__(256) void gemm_out(const __bf16* __restrict__ A,
                                                const float* __restrict__ W,
                                                const float* __restrict__ bo,
                                                float* __restrict__ out) {
  __shared__ __bf16 As[128 * LDSPAD];
  __shared__ __bf16 Bs[128 * LDSPAD];

  const int tid = threadIdx.x, lane = tid & 31, wave = tid >> 5;
  const int m0 = blockIdx.y * 128, n0 = blockIdx.x * 128;
  const int wm = (wave & 3) * 32, wn = (wave >> 2) * 64;
  const int rb = lane & 15, hi = lane >> 4;
  const int koff8 = hi * 8, koff16 = hi * 16;
  const int ar = tid >> 1, ak = (tid & 1) * 16;
  const int wk = tid >> 3, wn2 = (tid & 7) * 16;

  v8f acc[2][4] = {};

  for (int kk = 0; kk < 1024; kk += 32) {
    {
      const __bf16* ag = A + (size_t)(m0 + ar) * 1024 + kk + ak;
      *(uint4*)(As + ar * LDSPAD + ak)     = ((const uint4*)ag)[0];
      *(uint4*)(As + ar * LDSPAD + ak + 8) = ((const uint4*)ag)[1];
    }
    {
      const float* wg = W + (size_t)(kk + wk) * 1024 + n0 + wn2;
      float f[16];
#pragma unroll
      for (int j = 0; j < 4; ++j) {
        float4 t = ((const float4*)wg)[j];
        f[j * 4 + 0] = t.x; f[j * 4 + 1] = t.y; f[j * 4 + 2] = t.z; f[j * 4 + 3] = t.w;
      }
#pragma unroll
      for (int j = 0; j < 16; ++j) Bs[(wn2 + j) * LDSPAD + wk] = (__bf16)f[j];
    }
    __syncthreads();

    v16bf afr[2], bfr[4];
#pragma unroll
    for (int mi = 0; mi < 2; ++mi) {
      const __bf16* p = As + (wm + mi * 16 + rb) * LDSPAD;
      ((uint4*)&afr[mi])[0] = *(const uint4*)(p + koff16);
      ((uint4*)&afr[mi])[1] = *(const uint4*)(p + koff16 + 8);
    }
#pragma unroll
    for (int ni = 0; ni < 4; ++ni)
      ld_frag32B(bfr[ni], Bs + (wn + ni * 16 + rb) * LDSPAD + koff16);
#pragma unroll
    for (int mi = 0; mi < 2; ++mi)
#pragma unroll
      for (int ni = 0; ni < 4; ++ni)
        acc[mi][ni] = wmma_bf16(afr[mi], bfr[ni], acc[mi][ni]);
    __syncthreads();
  }

#pragma unroll
  for (int mi = 0; mi < 2; ++mi) {
#pragma unroll
    for (int ni = 0; ni < 4; ++ni) {
      const int mt = m0 + wm + mi * 16 + hi * 8;
      const int nt = n0 + wn + ni * 16 + rb;
      const float bias = bo[nt];
#pragma unroll
      for (int i = 0; i < 8; ++i)
        out[(size_t)(mt + i) * 1024 + nt] = acc[mi][ni][i] + bias;
    }
  }
}

// ---------------------------------------------------------------------------
extern "C" void kernel_launch(void* const* d_in, const int* in_sizes, int n_in,
                              void* d_out, int out_size, void* d_ws, size_t ws_size,
                              hipStream_t stream) {
  (void)in_sizes; (void)n_in; (void)out_size; (void)ws_size;
  const float* Xq = (const float*)d_in[0];
  const float* Xk = (const float*)d_in[1];
  const float* Xv = (const float*)d_in[2];
  const float* Wq = (const float*)d_in[5];
  const float* Wk = (const float*)d_in[6];
  const float* Wv = (const float*)d_in[7];
  const float* Wo = (const float*)d_in[8];
  const float* bo = (const float*)d_in[9];
  float* out = (float*)d_out;

  __bf16* Qb = (__bf16*)d_ws;
  __bf16* Kb = Qb + (size_t)4096 * 1024;
  __bf16* Vt = Kb + (size_t)4096 * 1024;
  __bf16* AO = Vt + (size_t)4096 * 1024;

  dim3 gGemm(8, 32);
  dim3 blk(256);
  const float qscale = 0.125f * 1.44269504f;  // HD^-0.5 * log2(e), folded into Q
  gemm_proj<false><<<gGemm, blk, 0, stream>>>(Xq, Wq, Qb, qscale);
  gemm_proj<false><<<gGemm, blk, 0, stream>>>(Xk, Wk, Kb, 1.0f);
  gemm_proj<true ><<<gGemm, blk, 0, stream>>>(Xv, Wv, Vt, 1.0f);

  dim3 gAttn(64, 8);
  attn_kernel<<<gAttn, blk, 0, stream>>>(Qb, Kb, Vt, AO, /*stage_mode=*/0);

  gemm_out<<<gGemm, blk, 0, stream>>>(AO, Wo, bo, out);
}
